// CustomGATLayer_25769804013
// MI455X (gfx1250) — matrix operand order repaired and checked by reading, not compiled, not run on hardware
//
#include <hip/hip_runtime.h>
#include <hip/hip_bf16.h>

#define N_NODES 4096
#define IN_FEAT 256
#define OUT_FEAT 256
#define NHEAD 4
#define FH 64
#define LRELU 0.2f

typedef __attribute__((ext_vector_type(2)))  float    v2f;
typedef __attribute__((ext_vector_type(8)))  float    v8f;
typedef __attribute__((ext_vector_type(8)))  _Float16 v8h;
typedef __attribute__((ext_vector_type(16))) _Float16 v16h;

// ---------------------------------------------------------------------------
// Kernel 1: hp = h @ W  via V_WMMA_F32_16X16X4_F32 (full fp32 tensor path).
// Each wave computes one 16x16 output tile, K-loop of 64 WMMAs (K=4 each).
// Epilogue writes hp (fp32, row-major) and hpT (f16, [channel][node]) so the
// attention kernel's WMMA B operands are contiguous per-lane loads.
// A layout (16x4 f32): lane L holds row M=L&15; VGPR0/1 = K = 2*(L>>4)+{0,1}.
// B layout (4x16 f32): lane L holds col N=L&15; VGPR0/1 = K = 2*(L>>4)+{0,1}.
// C layout: VGPR r -> row M = r + 8*(L>>4), col N = L&15.
// ---------------------------------------------------------------------------
__global__ __launch_bounds__(256) void gat_gemm_hp(
    const float* __restrict__ h, const float* __restrict__ W,
    float* __restrict__ hp, _Float16* __restrict__ hpT)
{
    const int wave = threadIdx.x >> 5;
    const int lane = threadIdx.x & 31;
    const int tile = blockIdx.x * 8 + wave;     // 4096 tiles total
    const int i0 = (tile >> 4) << 4;            // 256 row blocks
    const int n0 = (tile & 15) << 4;            // 16 col blocks
    const int row = lane & 15;                  // = col for B/C roles
    const int hi  = lane >> 4;

    const float* ap = h + (size_t)(i0 + row) * IN_FEAT + 2 * hi;
    const float* bp = W + (size_t)(2 * hi) * OUT_FEAT + n0 + row;

    v8f c = {};
#pragma unroll 4
    for (int k = 0; k < IN_FEAT; k += 4) {
        v2f av = *(const v2f*)(ap + k);
        v2f bv;
        bv[0] = bp[(size_t)k * OUT_FEAT];
        bv[1] = bp[(size_t)(k + 1) * OUT_FEAT];
        c = __builtin_amdgcn_wmma_f32_16x16x4_f32(
                false, av, false, bv, (short)0, c, false, false);
    }

    v8h tv;
#pragma unroll
    for (int r = 0; r < 8; ++r) {
        const int m = r + 8 * hi;
        hp[(size_t)(i0 + m) * OUT_FEAT + n0 + row] = c[r];
        tv[r] = (_Float16)c[r];
    }
    // hpT[channel][node]: per lane a contiguous 8-half (16 B) store.
    *(v8h*)(hpT + (size_t)(n0 + row) * N_NODES + i0 + 8 * hi) = tv;
}

// ---------------------------------------------------------------------------
// Kernel 2: s[h][n] = hp[n,h,:]·a_src[h],  d[h][n] = hp[n,h,:]·a_dst[h]
// ---------------------------------------------------------------------------
__global__ __launch_bounds__(256) void gat_sd(
    const float* __restrict__ hp, const float* __restrict__ a,
    float* __restrict__ s, float* __restrict__ d)
{
    const int idx = blockIdx.x * blockDim.x + threadIdx.x;   // 16384 = 4096*4
    const int n  = idx >> 2;
    const int hh = idx & 3;
    const float* v  = hp + (size_t)n * OUT_FEAT + hh * FH;
    const float* as = a + hh * 2 * FH;
    const float* ad = as + FH;
    float ss = 0.f, dd = 0.f;
#pragma unroll 8
    for (int f = 0; f < FH; ++f) {
        ss = fmaf(v[f], as[f], ss);
        dd = fmaf(v[f], ad[f], dd);
    }
    s[hh * N_NODES + n] = ss;
    d[hh * N_NODES + n] = dd;
}

// ---------------------------------------------------------------------------
// Kernel 3: flash-style masked softmax-aggregation.
// One wave per (head, 16-row block); loop over 32-wide column chunks.
// e is computed directly in the A operand order used here (same k order as B):
//   lane L owns row M=L&15; half t -> K = 16*hi + t
// so adj/d loads are one contiguous 16-float run per lane.
// B (32x16 f16): lane L holds col N=L&15, halves = hpT[col][j0 + 16*hi .. +15].
// ---------------------------------------------------------------------------
__global__ __launch_bounds__(128) void gat_attn(
    const float* __restrict__ adj, const _Float16* __restrict__ hpT,
    const float* __restrict__ s, const float* __restrict__ d,
    float* __restrict__ out)
{
    const int wave = threadIdx.x >> 5;
    const int lane = threadIdx.x & 31;
    const int head = blockIdx.y;
    const int i0   = blockIdx.x * 64 + wave * 16;
    const int row  = lane & 15;
    const int hi   = lane >> 4;
    const int koff = hi * 8;

    const float s_i = s[head * N_NODES + i0 + row];
    const float* __restrict__ dv     = d + head * N_NODES;
    const float* __restrict__ adjrow = adj + (size_t)(i0 + row) * N_NODES;
    const _Float16* __restrict__ b0p =
        hpT + (size_t)(head * FH + row) * N_NODES + hi * 16;

    v8f acc0 = {}, acc1 = {}, acc2 = {}, acc3 = {};
    float m_run = -3e38f;
    float l_run = 0.f;

    for (int j0 = 0; j0 < N_NODES; j0 += 32) {
        __builtin_prefetch(adjrow + j0 + 512, 0, 0);   // global_prefetch_b8

        const float4 a0 = *(const float4*)(adjrow + j0 + 2 * koff);
        const float4 a1 = *(const float4*)(adjrow + j0 + 2 * koff + 4);
        const float4 a2 = *(const float4*)(adjrow + j0 + 2 * koff + 8);
        const float4 a3 = *(const float4*)(adjrow + j0 + 2 * koff + 12);
        const float4 e0 = *(const float4*)(dv + j0 + 2 * koff);
        const float4 e1 = *(const float4*)(dv + j0 + 2 * koff + 4);
        const float4 e2 = *(const float4*)(dv + j0 + 2 * koff + 8);
        const float4 e3 = *(const float4*)(dv + j0 + 2 * koff + 12);
        const v16h b0 = *(const v16h*)(b0p + j0);
        const v16h b1 = *(const v16h*)(b0p + (size_t)16 * N_NODES + j0);
        const v16h b2 = *(const v16h*)(b0p + (size_t)32 * N_NODES + j0);
        const v16h b3 = *(const v16h*)(b0p + (size_t)48 * N_NODES + j0);

        const float aa[16] = {a0.x,a0.y,a0.z,a0.w, a1.x,a1.y,a1.z,a1.w,
                              a2.x,a2.y,a2.z,a2.w, a3.x,a3.y,a3.z,a3.w};
        const float dd[16] = {e0.x,e0.y,e0.z,e0.w, e1.x,e1.y,e1.z,e1.w,
                              e2.x,e2.y,e2.z,e2.w, e3.x,e3.y,e3.z,e3.w};

        float ev[16];
        float mloc = -3e38f;
#pragma unroll
        for (int t = 0; t < 16; ++t) {
            float e = s_i + dd[t];
            e = (e > 0.f) ? e : LRELU * e;
            ev[t] = (aa[t] > 0.f) ? e : -3e38f;
            mloc = fmaxf(mloc, ev[t]);
        }
        mloc = fmaxf(mloc, __shfl_xor(mloc, 16, 32));   // combine lane-pair rows
        const float m_new = fmaxf(m_run, mloc);
        const float alpha = __expf(m_run - m_new);
        m_run = m_new;

        float ls = 0.f;
        v16h ap;
#pragma unroll
        for (int t = 0; t < 16; ++t) {
            const float p = (aa[t] > 0.f) ? __expf(ev[t] - m_new) : 0.f;
            ls += p;
            ap[t] = (_Float16)p;
        }
        ls += __shfl_xor(ls, 16, 32);
        l_run = l_run * alpha + ls;

        // Rescale accumulators: acc VGPR r holds row (r + 8*hi); row stats
        // for rows 0..15 live in lanes 0..15.
#pragma unroll
        for (int r = 0; r < 8; ++r) {
            const float ar = __shfl(alpha, r + koff, 32);
            acc0[r] *= ar; acc1[r] *= ar; acc2[r] *= ar; acc3[r] *= ar;
        }

        acc0 = __builtin_amdgcn_wmma_f32_16x16x32_f16(false, ap, false, b0, (short)0, acc0, false, false);
        acc1 = __builtin_amdgcn_wmma_f32_16x16x32_f16(false, ap, false, b1, (short)0, acc1, false, false);
        acc2 = __builtin_amdgcn_wmma_f32_16x16x32_f16(false, ap, false, b2, (short)0, acc2, false, false);
        acc3 = __builtin_amdgcn_wmma_f32_16x16x32_f16(false, ap, false, b3, (short)0, acc3, false, false);
    }

    float* __restrict__ ocol = out + (size_t)head * FH;
#pragma unroll
    for (int r = 0; r < 8; ++r) {
        const float lr  = __shfl(l_run, r + koff, 32);
        const float inv = 1.f / lr;
        const int m = i0 + r + 8 * hi;
        const float vals[4] = {acc0[r]*inv, acc1[r]*inv, acc2[r]*inv, acc3[r]*inv};
#pragma unroll
        for (int nt = 0; nt < 4; ++nt) {
            float x = vals[nt];
            x = (x > 0.f) ? x : (__expf(x) - 1.f);      // ELU
            ocol[(size_t)m * OUT_FEAT + nt * 16 + row] = x;
        }
    }
}

// ---------------------------------------------------------------------------
extern "C" void kernel_launch(void* const* d_in, const int* in_sizes, int n_in,
                              void* d_out, int out_size, void* d_ws, size_t ws_size,
                              hipStream_t stream) {
    const float* h   = (const float*)d_in[0];
    const float* adj = (const float*)d_in[1];
    const float* W   = (const float*)d_in[2];
    const float* a   = (const float*)d_in[3];
    float* out = (float*)d_out;

    char* ws = (char*)d_ws;
    float*    hp  = (float*)ws;                                   // 4 MB
    _Float16* hpT = (_Float16*)(ws + (size_t)4 * 1024 * 1024);    // 2 MB
    float*    s   = (float*)(ws + (size_t)6 * 1024 * 1024);       // 64 KB
    float*    d   = (float*)(ws + (size_t)6 * 1024 * 1024 + 64 * 1024);

    hipLaunchKernelGGL(gat_gemm_hp, dim3(512),   dim3(256), 0, stream, h, W, hp, hpT);
    hipLaunchKernelGGL(gat_sd,      dim3(64),    dim3(256), 0, stream, hp, a, s, d);
    hipLaunchKernelGGL(gat_attn,    dim3(64, 4), dim3(128), 0, stream, adj, hpT, s, d, out);
}
